// AAS_38130719653971
// MI455X (gfx1250) — matrix operand order repaired and checked by reading, not compiled, not run on hardware
//
#include <hip/hip_runtime.h>

#define PD_EPS 1e-6f

typedef __attribute__((ext_vector_type(16))) __bf16 v16bf;
typedef __attribute__((ext_vector_type(8)))  __bf16 v8bf;
typedef __attribute__((ext_vector_type(2)))  __bf16 v2bf;
typedef __attribute__((ext_vector_type(8)))  float  v8f;

// ---------------------------------------------------------------- utilities

// native packed f32 -> bf16 convert (lowers to v_cvt_pk_bf16_f32 on gfx1250)
static __device__ __forceinline__ unsigned pack_bf16(float a, float b) {
    v2bf p;
    p.x = (__bf16)a;
    p.y = (__bf16)b;
    return __builtin_bit_cast(unsigned, p);
}

static __device__ __forceinline__ float wave_sum(float x) {
    #pragma unroll
    for (int off = 16; off > 0; off >>= 1) x += __shfl_xor(x, off, 32);
    return x;
}

static __device__ __forceinline__ float wave_max(float x) {
    #pragma unroll
    for (int off = 16; off > 0; off >>= 1) x = fmaxf(x, __shfl_xor(x, off, 32));
    return x;
}

// ------------------------------------------------- per-row sums (fp32 exact)
// one wave32 per row; block = 8 waves

__global__ __launch_bounds__(256)
void row_stats_kernel(const float* __restrict__ X, int nrows, int D,
                      float* __restrict__ sq, float* __restrict__ sm) {
    const int wave = threadIdx.x >> 5;
    const int lane = threadIdx.x & 31;
    const int row  = blockIdx.x * 8 + wave;
    if (row >= nrows) return;
    const float4* p = (const float4*)(X + (size_t)row * D);
    const int nv = D >> 2;
    float s2 = 0.0f, s1 = 0.0f;
    for (int i = lane; i < nv; i += 32) {
        float4 v = p[i];
        s2 += v.x * v.x + v.y * v.y + v.z * v.z + v.w * v.w;
        s1 += v.x + v.y + v.z + v.w;
    }
    s2 = wave_sum(s2);
    s1 = wave_sum(s1);
    if (lane == 0) { sq[row] = s2; sm[row] = s1; }
}

// ------------------------------------------------- bf16 WMMA distance GEMM
// Block tile: 128(Q) x 128(C), K-tile 64. 256 threads = 8 waves in 4x2 grid;
// each wave computes a 32x64 block = 2x4 v_wmma_f32_16x16x32_bf16 tiles.
// Grid: x = column tiles (C/128 = 8, fastest-varying) so the 8 blocks covering
// one 128-row query panel are co-resident -> queries stream from HBM once,
// protos (8 MB) stay L2-resident.

#define BM  128
#define BN  128
#define BK  64
#define LDP 72   // LDS pitch in bf16 elements (144B rows -> conflict-free b128)

__global__ __launch_bounds__(256)
void dist_gemm_kernel(const float* __restrict__ Qm, const float* __restrict__ Pm,
                      const float* __restrict__ q2, const float* __restrict__ qs,
                      const float* __restrict__ p2, const float* __restrict__ ps,
                      float* __restrict__ U, int Crows, int D) {
    __shared__ __align__(16) unsigned short As[BM * LDP];
    __shared__ __align__(16) unsigned short Bs[BN * LDP];

    const int tid  = threadIdx.x;
    const int wave = tid >> 5;
    const int lane = tid & 31;
    const int half = lane >> 4;   // which 16-lane half
    const int l16  = lane & 15;

    const int wm = wave >> 1;     // 0..3 -> 32-row slab
    const int wn = wave & 1;      // 0..1 -> 64-col slab
    const int tileC = blockIdx.x * BN;   // column tile (fast dim for L2 reuse)
    const int tileQ = blockIdx.y * BM;   // query row panel

    v8f acc[2][4];
    const v8f vzero = {0.f, 0.f, 0.f, 0.f, 0.f, 0.f, 0.f, 0.f};
    #pragma unroll
    for (int i = 0; i < 2; ++i)
        #pragma unroll
        for (int j = 0; j < 4; ++j) acc[i][j] = vzero;

    // global -> LDS staging map: 2 threads per tile row, 32 fp32 each
    const int ldRow  = tid >> 1;      // 0..127
    const int ldHalf = tid & 1;       // k offset 0 / 32

    for (int kt = 0; kt < D; kt += BK) {
        { // stage A (queries) tile, fp32 -> bf16 via packed hardware convert
            const float4* src = (const float4*)(Qm + (size_t)(tileQ + ldRow) * D
                                                + kt + ldHalf * 32);
            unsigned short* dst = As + ldRow * LDP + ldHalf * 32;
            #pragma unroll
            for (int i = 0; i < 8; ++i) {
                float4 v = src[i];
                uint2 pk;
                pk.x = pack_bf16(v.x, v.y);
                pk.y = pack_bf16(v.z, v.w);
                *(uint2*)(dst + i * 4) = pk;
            }
        }
        { // stage B (protos) tile
            const float4* src = (const float4*)(Pm + (size_t)(tileC + ldRow) * D
                                                + kt + ldHalf * 32);
            unsigned short* dst = Bs + ldRow * LDP + ldHalf * 32;
            #pragma unroll
            for (int i = 0; i < 8; ++i) {
                float4 v = src[i];
                uint2 pk;
                pk.x = pack_bf16(v.x, v.y);
                pk.y = pack_bf16(v.z, v.w);
                *(uint2*)(dst + i * 4) = pk;
            }
        }
        __syncthreads();

        #pragma unroll
        for (int kk = 0; kk < BK; kk += 32) {
            // Hoist ALL fragment loads (12x ds_load_b128) ahead of the WMMA
            // burst so a single s_wait_dscnt covers the 8 matrix ops.
            v16bf afrag[2];
            v16bf bfrag[4];
            #pragma unroll
            for (int i = 0; i < 2; ++i) {
                // A 16x32 bf16: lane(half,m) holds K chunks [half*8..+7], [16+half*8..+7]
                const unsigned short* base =
                    As + (wm * 32 + i * 16 + l16) * LDP + kk;
                v8bf lo = *(const v8bf*)(base + half * 8);
                v8bf hi = *(const v8bf*)(base + 16 + half * 8);
                afrag[i] = __builtin_shufflevector(lo, hi,
                    0,1,2,3,4,5,6,7,8,9,10,11,12,13,14,15);
            }
            #pragma unroll
            for (int j = 0; j < 4; ++j) {
                // B 32x16 bf16: lane(half,n) holds K chunks [half*8..+7], [16+half*8..+7] of col n
                const unsigned short* base =
                    Bs + (wn * 64 + j * 16 + l16) * LDP + kk + half * 8;
                v8bf lo = *(const v8bf*)(base);
                v8bf hi = *(const v8bf*)(base + 16);
                bfrag[j] = __builtin_shufflevector(lo, hi,
                    0,1,2,3,4,5,6,7,8,9,10,11,12,13,14,15);
            }
            #pragma unroll
            for (int j = 0; j < 4; ++j) {
                #pragma unroll
                for (int i = 0; i < 2; ++i) {
                    acc[i][j] = __builtin_amdgcn_wmma_f32_16x16x32_bf16(
                        false, afrag[i], false, bfrag[j],
                        (short)0, acc[i][j], false, false);
                }
            }
        }
        __syncthreads();
    }

    // epilogue: sqd = q2 + p2 - 2*cross + 2*eps*(qs - ps) + D*eps^2 ; u = exp(-sqrt(sqd))
    const float corrD = (float)D * PD_EPS * PD_EPS;
    #pragma unroll
    for (int j = 0; j < 4; ++j) {
        const int col = tileC + wn * 64 + j * 16 + l16;
        const float pc2 = p2[col];
        const float pcs = ps[col];
        #pragma unroll
        for (int i = 0; i < 2; ++i) {
            #pragma unroll
            for (int r = 0; r < 8; ++r) {
                const int row = tileQ + wm * 32 + i * 16 + half * 8 + r;
                const float x = acc[i][j][r];
                float sqd = q2[row] + pc2 - 2.0f * x
                          + 2.0f * PD_EPS * (qs[row] - pcs) + corrD;
                sqd = fmaxf(sqd, 0.0f);
                U[(size_t)row * Crows + col] = __expf(-sqrtf(sqd));
            }
        }
    }
}

// ------------------------------------- row softmax-normalize + conf + entropy
// one wave32 per row of C elements (in-place on U)

__global__ __launch_bounds__(256)
void posterior_rows_kernel(float* __restrict__ U, float* __restrict__ conf,
                           float* __restrict__ ent, int Qrows, int Crows) {
    const int wave = threadIdx.x >> 5;
    const int lane = threadIdx.x & 31;
    const int row  = blockIdx.x * 8 + wave;
    if (row >= Qrows) return;
    float4* rp = (float4*)(U + (size_t)row * Crows);
    const int nv = Crows >> 2;

    float s = 0.0f, m = -3.4e38f;
    for (int i = lane; i < nv; i += 32) {
        float4 u = rp[i];
        s += u.x + u.y + u.z + u.w;
        m = fmaxf(m, fmaxf(fmaxf(u.x, u.y), fmaxf(u.z, u.w)));
    }
    s = wave_sum(s);
    m = wave_max(m);
    const float inv = 1.0f / s;
    const float lgS = __logf(s);

    float h = 0.0f;
    for (int i = lane; i < nv; i += 32) {
        float4 u = rp[i];
        float4 p;
        p.x = u.x * inv; h -= p.x * (__logf(u.x) - lgS);
        p.y = u.y * inv; h -= p.y * (__logf(u.y) - lgS);
        p.z = u.z * inv; h -= p.z * (__logf(u.z) - lgS);
        p.w = u.w * inv; h -= p.w * (__logf(u.w) - lgS);
        rp[i] = p;
    }
    h = wave_sum(h);
    if (lane == 0) { conf[row] = m * inv; ent[row] = h; }
}

// ---------------------------------------------------------------- selection

__global__ void select_mask_kernel(const float* __restrict__ cr,
                                   const float* __restrict__ hr,
                                   const float* __restrict__ cf,
                                   const float* __restrict__ hf,
                                   float* __restrict__ sel, int Q) {
    const int i = blockIdx.x * blockDim.x + threadIdx.x;
    if (i < Q) sel[i] = ((hr[i] > hf[i]) && (cr[i] > cf[i])) ? 1.0f : 0.0f;
}

// ------------------------------------------------------------------ launcher

extern "C" void kernel_launch(void* const* d_in, const int* in_sizes, int n_in,
                              void* d_out, int out_size, void* d_ws, size_t ws_size,
                              hipStream_t stream) {
    const int D = 2048;
    const float* Pr = (const float*)d_in[0];   // context_rgb  [C,D]
    const float* Pf = (const float*)d_in[1];   // context_flow [C,D]
    const float* Qr = (const float*)d_in[2];   // target_rgb   [Q,D]
    const float* Qf = (const float*)d_in[3];   // target_flow  [Q,D]
    const int C = in_sizes[0] / D;
    const int Q = in_sizes[2] / D;
    const size_t QC = (size_t)Q * (size_t)C;

    float* out = (float*)d_out;
    float* p_r = out;
    float* c_r = p_r + QC;
    float* h_r = c_r + Q;
    float* p_f = h_r + Q;
    float* c_f = p_f + QC;
    float* h_f = c_f + Q;
    float* sel = h_f + Q;

    float* ws  = (float*)d_ws;
    float* q2r = ws;        float* qsr = q2r + Q;
    float* q2f = qsr + Q;   float* qsf = q2f + Q;
    float* p2r = qsf + Q;   float* psr = p2r + C;
    float* p2f = psr + C;   float* psf = p2f + C;

    const dim3 blk(256);

    row_stats_kernel<<<(Q + 7) / 8, blk, 0, stream>>>(Qr, Q, D, q2r, qsr);
    row_stats_kernel<<<(Q + 7) / 8, blk, 0, stream>>>(Qf, Q, D, q2f, qsf);
    row_stats_kernel<<<(C + 7) / 8, blk, 0, stream>>>(Pr, C, D, p2r, psr);
    row_stats_kernel<<<(C + 7) / 8, blk, 0, stream>>>(Pf, C, D, p2f, psf);

    const dim3 grid(C / BN, Q / BM);   // x = column tiles (fast) for L2 reuse
    dist_gemm_kernel<<<grid, blk, 0, stream>>>(Qr, Pr, q2r, qsr, p2r, psr, p_r, C, D);
    dist_gemm_kernel<<<grid, blk, 0, stream>>>(Qf, Pf, q2f, qsf, p2f, psf, p_f, C, D);

    posterior_rows_kernel<<<(Q + 7) / 8, blk, 0, stream>>>(p_r, c_r, h_r, Q, C);
    posterior_rows_kernel<<<(Q + 7) / 8, blk, 0, stream>>>(p_f, c_f, h_f, Q, C);

    select_mask_kernel<<<(Q + 255) / 256, blk, 0, stream>>>(c_r, h_r, c_f, h_f, sel, Q);
}
